// Transformer_49495203119478
// MI455X (gfx1250) — matrix operand order repaired and checked by reading, not compiled, not run on hardware
//
#include <hip/hip_runtime.h>

// ---------------------------------------------------------------------------
// CDNA5 (gfx1250) transformer forward: WMMA f16 (f32 acc) + async LDS staging.
// ---------------------------------------------------------------------------

typedef __attribute__((ext_vector_type(16))) _Float16 v16h;
typedef __attribute__((ext_vector_type(8)))  float    v8f;
typedef __attribute__((ext_vector_type(4)))  int      v4i;

#define TM 128
#define TK 32
#define LS 48   // LDS row stride in halves; 96B rows -> 16B aligned sub-chunks

union Frag16 { uint4 u[2]; v16h h; };

// ---- CDNA5 async memory->LDS copy (ASYNCcnt) with safe fallback ----
#if defined(__has_builtin)
#  if __has_builtin(__builtin_amdgcn_global_load_async_to_lds_b128)
#    define USE_ASYNC_LDS 1
#  endif
#  if __has_builtin(__builtin_amdgcn_s_wait_asynccnt)
#    define HAVE_WAIT_ASYNC 1
#  endif
#endif

typedef __attribute__((address_space(1))) v4i GV4;   // global 16B chunk
typedef __attribute__((address_space(3))) v4i LV4;   // LDS    16B chunk

__device__ __forceinline__ void cp16(const _Float16* g, _Float16* l) {
#ifdef USE_ASYNC_LDS
  __builtin_amdgcn_global_load_async_to_lds_b128((GV4*)g, (LV4*)l, 0, 0);
#else
  *(uint4*)l = *(const uint4*)g;
#endif
}

template <int N>
__device__ __forceinline__ void wait_cp() {
#ifdef USE_ASYNC_LDS
#  ifdef HAVE_WAIT_ASYNC
  __builtin_amdgcn_s_wait_asynccnt(N);
#  else
  asm volatile("s_wait_asynccnt %0" :: "i"(N) : "memory");
#  endif
#endif
}

// Generic batched GEMM:  C[z] = act( scale * A[z] x B[z]^T + bias ).
// A: (M x K) row-major f16.  B: (N x K) row-major f16 (i.e. B^T of math B).
// ct=0: C[m][n] at m*ldc+n.  ct=1: store transposed, C[n][m] at n*ldc+m.
// Per-z offsets: off = (z/zdiv)*Outer + (z%zdiv)*Inner (elements).
// M % 128 == 0, N % TNT == 0, K % 32 == 0.
template <int TNT>
__global__ __launch_bounds__(256)
void gemm_wmma_kernel(const _Float16* __restrict__ A, long long lda,
                      const _Float16* __restrict__ B, long long ldb,
                      const float* __restrict__ bias,
                      float* C32, _Float16* C16, long long ldc, int ct,
                      int M, int N, int K, float scale, int relu,
                      int zdiv, long long aO, long long aI,
                      long long bO, long long bI,
                      long long cO, long long cI)
{
  constexpr int NI = TNT / 32;           // 16-wide WMMA subtiles per wave (N)
  constexpr int NB = (TNT * 4) / 256;    // B staging chunks per thread
  constexpr int TI = 2 + NB;             // async issues per tile per thread
  __shared__ __align__(16) _Float16 AsBuf[2 * TM * LS];
  __shared__ __align__(16) _Float16 BsBuf[2 * TNT * LS];

  const int z = blockIdx.z;
  A += (long long)(z / zdiv) * aO + (long long)(z % zdiv) * aI;
  B += (long long)(z / zdiv) * bO + (long long)(z % zdiv) * bI;
  const long long coff = (long long)(z / zdiv) * cO + (long long)(z % zdiv) * cI;

  const int tid  = threadIdx.x;
  const int lane = tid & 31;
  const int wave = tid >> 5;             // 8 waves: 4 (M) x 2 (N)
  const int wm   = wave & 3;             // 32-row strip
  const int wn   = wave >> 2;            // (TNT/2)-col strip
  const int lh   = lane & 15;
  const int sel  = lane >> 4;            // half-wave select (WMMA VGPR layout)
  const int m0   = blockIdx.x * TM;
  const int n0   = blockIdx.y * TNT;

  // stage one (TM x TK) A tile + (TNT x TK) B^T tile into buffer `buf`
  auto stage = [&](int k0, int buf) {
    _Float16* As = AsBuf + buf * (TM * LS);
    _Float16* Bs = BsBuf + buf * (TNT * LS);
#pragma unroll
    for (int it = 0; it < 2; ++it) {               // A: 512 chunks / 256 thr
      int id = tid + it * 256;
      int r = id >> 2, kc = id & 3;
      cp16(A + (long long)(m0 + r) * lda + k0 + kc * 8, &As[r * LS + kc * 8]);
    }
#pragma unroll
    for (int it = 0; it < NB; ++it) {              // B: TNT*4 chunks
      int id = tid + it * 256;
      int n = id >> 2, kc = id & 3;
      cp16(B + (long long)(n0 + n) * ldb + k0 + kc * 8, &Bs[n * LS + kc * 8]);
    }
  };

  v8f acc[2][NI] = {};

  stage(0, 0);
  for (int k0 = 0; k0 < K; k0 += TK) {
    const int cur = (k0 / TK) & 1;
    if (k0 + TK < K) {      // prefetch next tile into other buffer, wait cur
      stage(k0 + TK, cur ^ 1);
      wait_cp<TI>();
    } else {
      wait_cp<0>();
    }
    __syncthreads();

    const _Float16* As = AsBuf + cur * (TM * LS);
    const _Float16* Bs = BsBuf + cur * (TNT * LS);

    // ---- build fragments (ISA 16-bit A 16x32 / B 32x16 wave32 layouts)
    Frag16 fa[2], fb[NI];
#pragma unroll
    for (int mi = 0; mi < 2; ++mi) {
      const _Float16* ar = &As[(wm * 32 + mi * 16 + lh) * LS + sel * 8];
      fa[mi].u[0] = *(const uint4*)ar;        // K 0..7  (or 8..15)
      fa[mi].u[1] = *(const uint4*)(ar + 16); // K 16..23 (or 24..31)
    }
#pragma unroll
    for (int ni = 0; ni < NI; ++ni) {
      const _Float16* br = &Bs[(wn * (16 * NI) + ni * 16 + lh) * LS + sel * 8];
      fb[ni].u[0] = *(const uint4*)br;        // K 0..7  (or 8..15)
      fb[ni].u[1] = *(const uint4*)(br + 16); // K 16..23 (or 24..31)
    }

#pragma unroll
    for (int mi = 0; mi < 2; ++mi)
#pragma unroll
      for (int ni = 0; ni < NI; ++ni)
        acc[mi][ni] = __builtin_amdgcn_wmma_f32_16x16x32_f16(
            false, fa[mi].h, false, fb[ni].h,
            (short)0, acc[mi][ni], false, false);
    __syncthreads();
  }

  // ---- epilogue: scale + bias + relu, normal or transposed stores
#pragma unroll
  for (int mi = 0; mi < 2; ++mi)
#pragma unroll
    for (int ni = 0; ni < NI; ++ni) {
      int n = n0 + wn * (16 * NI) + ni * 16 + lh;
      float bv = bias ? bias[n] : 0.f;
#pragma unroll
      for (int r = 0; r < 8; ++r) {
        int m = m0 + wm * 32 + mi * 16 + sel * 8 + r;  // C/D VGPR layout
        float v = acc[mi][ni][r] * scale + bv;
        if (relu && v < 0.f) v = 0.f;
        long long idx = coff + (ct ? (long long)n * ldc + m
                                   : (long long)m * ldc + n);
        if (C32) C32[idx] = v;
        if (C16) C16[idx] = (_Float16)v;
      }
    }
}

// ---- input embedding: interleave ctx / one-hot labels + one-hot position ----
__global__ void embed_kernel(const float* __restrict__ stim,
                             const int* __restrict__ labels,
                             float* __restrict__ X32, _Float16* __restrict__ X16)
{
  int row = blockIdx.x;            // b*512 + s, s in [0,512) (511 = pad row)
  int b = row >> 9, s = row & 511;
  int t = threadIdx.x;
#pragma unroll
  for (int i = 0; i < 4; ++i) {
    int c = t + i * 256;
    float v = 0.f;
    if (s < 511) {
      if (c < 512) {
        if (s == 510)            v = stim[((long long)b * 256 + 255) * 512 + c];
        else if ((s & 1) == 0)   v = stim[((long long)b * 256 + (s >> 1)) * 512 + c];
        else                     v = (labels[b * 255 + (s >> 1)] == c) ? 1.f : 0.f;
      } else {
        v = ((c - 512) == s) ? 1.f : 0.f;
      }
    }
    long long idx = (long long)row * 1024 + c;
    X32[idx] = v;
    X16[idx] = (_Float16)v;
  }
}

// ---- fp32 (K x N) -> fp16 transposed (N x K), per layer z; 32x32 LDS tiles ----
__global__ void cvt_t_kernel(const float* __restrict__ src,
                             _Float16* __restrict__ dst, int K, int N)
{
  __shared__ float tile[32][33];
  long long zoff = (long long)blockIdx.z * K * N;
  src += zoff;
  dst += zoff;
  int k0 = blockIdx.y * 32, n0 = blockIdx.x * 32;
  int tx = threadIdx.x, ty = threadIdx.y;     // (32, 8)
#pragma unroll
  for (int i = 0; i < 32; i += 8)
    tile[ty + i][tx] = src[(long long)(k0 + ty + i) * N + n0 + tx];
  __syncthreads();
#pragma unroll
  for (int i = 0; i < 32; i += 8)
    dst[(long long)(n0 + ty + i) * K + k0 + tx] = (_Float16)tile[tx][ty + i];
}

// ---- LayerNorm (+optional residual), in-place capable, dual output ----
__global__ __launch_bounds__(256)
void ln_kernel(const float* x, const float* res,
               const float* __restrict__ gam, const float* __restrict__ bet,
               float* o32, _Float16* o16)
{
  __shared__ float red[256];
  int row = blockIdx.x, t = threadIdx.x;
  float v[4];
#pragma unroll
  for (int i = 0; i < 4; ++i) {
    long long idx = (long long)row * 1024 + t + i * 256;
    v[i] = x[idx] + (res ? res[idx] : 0.f);
  }
  float s = v[0] + v[1] + v[2] + v[3];
  red[t] = s; __syncthreads();
  for (int o = 128; o > 0; o >>= 1) { if (t < o) red[t] += red[t + o]; __syncthreads(); }
  float mu = red[0] * (1.f / 1024.f);
  __syncthreads();
  float vs = 0.f;
#pragma unroll
  for (int i = 0; i < 4; ++i) { float d = v[i] - mu; vs += d * d; }
  red[t] = vs; __syncthreads();
  for (int o = 128; o > 0; o >>= 1) { if (t < o) red[t] += red[t + o]; __syncthreads(); }
  float r = rsqrtf(red[0] * (1.f / 1024.f) + 1e-5f);
#pragma unroll
  for (int i = 0; i < 4; ++i) {
    int c = t + i * 256;
    long long idx = (long long)row * 1024 + c;
    float y = (v[i] - mu) * r * gam[c] + bet[c];
    o32[idx] = y;
    if (o16) o16[idx] = (_Float16)y;
  }
}

// ---- row softmax over 512 keys, masking padded key 511; outputs f16 ----
__global__ __launch_bounds__(256)
void softmax_kernel(const float* __restrict__ S, _Float16* __restrict__ Aout)
{
  __shared__ float red[256];
  long long row = blockIdx.x;        // (b*16+h)*512 + q
  int t = threadIdx.x;
  float v0 = S[row * 512 + t];
  float v1 = S[row * 512 + t + 256];
  if (t == 255) v1 = -3.0e38f;       // key 511 is the pad row
  float m = fmaxf(v0, v1);
  red[t] = m; __syncthreads();
  for (int o = 128; o > 0; o >>= 1) { if (t < o) red[t] = fmaxf(red[t], red[t + o]); __syncthreads(); }
  float mx = red[0]; __syncthreads();
  float e0 = __expf(v0 - mx);
  float e1 = (t == 255) ? 0.f : __expf(v1 - mx);
  red[t] = e0 + e1; __syncthreads();
  for (int o = 128; o > 0; o >>= 1) { if (t < o) red[t] += red[t + o]; __syncthreads(); }
  float inv = 1.f / red[0];
  Aout[row * 512 + t]       = (_Float16)(e0 * inv);
  Aout[row * 512 + t + 256] = (_Float16)(e1 * inv);
}

// ---- tiny head: only the 8 last-token rows are needed ----
__global__ void head_kernel(const float* __restrict__ X, const float* __restrict__ Wh,
                            const float* __restrict__ bh, float* __restrict__ out)
{
  int b = blockIdx.x;                         // 0..7
  int n = blockIdx.y * 256 + threadIdx.x;     // 0..511
  const float* xr = X + ((long long)b * 512 + 510) * 1024;
  float s = bh[n];
  for (int k = 0; k < 1024; ++k) s += xr[k] * Wh[(long long)k * 512 + n];
  out[b * 512 + n] = s;
}

// ---------------------------------------------------------------------------
extern "C" void kernel_launch(void* const* d_in, const int* in_sizes, int n_in,
                              void* d_out, int out_size, void* d_ws, size_t ws_size,
                              hipStream_t stream)
{
  (void)in_sizes; (void)n_in; (void)out_size; (void)ws_size;
  const float* stim  = (const float*)d_in[0];
  const int*   lab   = (const int*)  d_in[1];
  const float* Wq    = (const float*)d_in[2];
  const float* bq    = (const float*)d_in[3];
  const float* Wk    = (const float*)d_in[4];
  const float* bk    = (const float*)d_in[5];
  const float* Wv    = (const float*)d_in[6];
  const float* bv    = (const float*)d_in[7];
  const float* Wo    = (const float*)d_in[8];
  const float* bo    = (const float*)d_in[9];
  const float* ln1s  = (const float*)d_in[10];
  const float* ln1b  = (const float*)d_in[11];
  const float* W1    = (const float*)d_in[12];
  const float* b1    = (const float*)d_in[13];
  const float* W2    = (const float*)d_in[14];
  const float* b2    = (const float*)d_in[15];
  const float* ln2s  = (const float*)d_in[16];
  const float* ln2b  = (const float*)d_in[17];
  const float* lnfs  = (const float*)d_in[18];
  const float* lnfb  = (const float*)d_in[19];
  const float* Wh    = (const float*)d_in[20];
  const float* bh    = (const float*)d_in[21];
  float* out = (float*)d_out;

  // ---- workspace carve-out ----
  char* p = (char*)d_ws;
  auto alloc = [&](size_t bytes) -> void* {
    void* r = (void*)p;
    p += (bytes + 255) & ~(size_t)255;
    return r;
  };
  const long long HW = 1024LL * 1024;        // h x h elems per layer
  const long long HM = 1024LL * 4096;        // h x mlp elems per layer
  // all weight copies are TRANSPOSED to (N x K) f16
  _Float16* Wq16 = (_Float16*)alloc(8 * HW * 2);
  _Float16* Wk16 = (_Float16*)alloc(8 * HW * 2);
  _Float16* Wv16 = (_Float16*)alloc(8 * HW * 2);
  _Float16* Wo16 = (_Float16*)alloc(8 * HW * 2);
  _Float16* W116 = (_Float16*)alloc(8 * HM * 2);
  _Float16* W216 = (_Float16*)alloc(8 * HM * 2);
  float*    X32  = (float*)   alloc(4096LL * 1024 * 4);
  _Float16* Xh   = (_Float16*)alloc(4096LL * 1024 * 2);
  _Float16* Qh   = (_Float16*)alloc(4096LL * 1024 * 2);
  _Float16* Kh   = (_Float16*)alloc(4096LL * 1024 * 2);
  _Float16* Vt   = (_Float16*)alloc(4096LL * 1024 * 2);  // V^T: [1024][4096]
  float*    Sc   = (float*)   alloc(128LL * 512 * 512 * 4);
  _Float16* Ah   = (_Float16*)alloc(128LL * 512 * 512 * 2);
  _Float16* Oh   = (_Float16*)alloc(4096LL * 1024 * 2);
  float*    Tmp  = (float*)   alloc(4096LL * 1024 * 4);
  _Float16* H1h  = (_Float16*)alloc(4096LL * 4096 * 2);

  // transpose+convert weights: src [L][K][N] f32 -> dst [L][N][K] f16
  auto cvtT = [&](const float* s, _Float16* d, int K, int N, int L) {
    dim3 g(N / 32, K / 32, L), b(32, 8);
    cvt_t_kernel<<<g, b, 0, stream>>>(s, d, K, N);
  };
  cvtT(Wq, Wq16, 1024, 1024, 8);
  cvtT(Wk, Wk16, 1024, 1024, 8);
  cvtT(Wv, Wv16, 1024, 1024, 8);
  cvtT(Wo, Wo16, 1024, 1024, 8);
  cvtT(W1, W116, 1024, 4096, 8);
  cvtT(W2, W216, 4096, 1024, 8);

  embed_kernel<<<dim3(4096), dim3(256), 0, stream>>>(stim, lab, X32, Xh);

  auto gemm = [&](int tn,
                  const _Float16* A, long long lda, const _Float16* Bp, long long ldb,
                  const float* bias, float* C32, _Float16* C16, long long ldc, int ct,
                  int M, int N, int K, float scale, int relu,
                  int z, int zdiv,
                  long long aO, long long aI, long long bO, long long bI,
                  long long cO, long long cI) {
    dim3 g(M / TM, N / tn, z), blk(256, 1, 1);
    if (tn == 128)
      gemm_wmma_kernel<128><<<g, blk, 0, stream>>>(A, lda, Bp, ldb, bias, C32, C16,
          ldc, ct, M, N, K, scale, relu, zdiv, aO, aI, bO, bI, cO, cI);
    else
      gemm_wmma_kernel<64><<<g, blk, 0, stream>>>(A, lda, Bp, ldb, bias, C32, C16,
          ldc, ct, M, N, K, scale, relu, zdiv, aO, aI, bO, bI, cO, cI);
  };

  const long long BS = 512LL * 1024;   // per-batch activation stride (elems)
  const long long SS = 512LL * 512;    // per-(b,head) score stride (elems)

  for (int L = 0; L < 8; ++L) {
    const _Float16* wq = Wq16 + (long long)L * HW;
    const _Float16* wk = Wk16 + (long long)L * HW;
    const _Float16* wv = Wv16 + (long long)L * HW;
    const _Float16* wo = Wo16 + (long long)L * HW;
    const _Float16* w1 = W116 + (long long)L * HM;
    const _Float16* w2 = W216 + (long long)L * HM;

    // QKV projections: (4096x1024) x (1024x1024);  V written transposed
    gemm(128, Xh, 1024, wq, 1024, bq + L * 1024, nullptr, Qh, 1024, 0,
         4096, 1024, 1024, 1.f, 0, 1, 1, 0, 0, 0, 0, 0, 0);
    gemm(128, Xh, 1024, wk, 1024, bk + L * 1024, nullptr, Kh, 1024, 0,
         4096, 1024, 1024, 1.f, 0, 1, 1, 0, 0, 0, 0, 0, 0);
    gemm(128, Xh, 1024, wv, 1024, bv + L * 1024, nullptr, Vt, 4096, /*ct=*/1,
         4096, 1024, 1024, 1.f, 0, 1, 1, 0, 0, 0, 0, 0, 0);

    // scores = Q K^T / 8 per (b,head): z=128, b=z/16, h=z%16
    gemm(128, Qh, 1024, Kh, 1024, nullptr, Sc, nullptr, 512, 0,
         512, 512, 64, 0.125f, 0,
         128, 16, BS, 64, BS, 64, 16 * SS, SS);

    softmax_kernel<<<dim3(128 * 512), dim3(256), 0, stream>>>(Sc, Ah);

    // O = A V per (b,head): A (512x512), B = V^T slice (64 x 512)
    gemm(64, Ah, 512, Vt, 4096, nullptr, nullptr, Oh, 1024, 0,
         512, 64, 512, 1.f, 0,
         128, 16, 16 * SS, SS, 512, 64LL * 4096, BS, 64);

    // output projection + residual + LN1
    gemm(128, Oh, 1024, wo, 1024, bo + L * 1024, Tmp, nullptr, 1024, 0,
         4096, 1024, 1024, 1.f, 0, 1, 1, 0, 0, 0, 0, 0, 0);
    ln_kernel<<<dim3(4096), dim3(256), 0, stream>>>(X32, Tmp, ln1s + L * 1024,
                                                    ln1b + L * 1024, X32, Xh);

    // MLP: relu(X W1 + b1) W2 + b2, residual + LN2
    gemm(128, Xh, 1024, w1, 1024, b1 + L * 4096, nullptr, H1h, 4096, 0,
         4096, 4096, 1024, 1.f, 1, 1, 1, 0, 0, 0, 0, 0, 0);
    gemm(128, H1h, 4096, w2, 4096, b2 + L * 1024, Tmp, nullptr, 1024, 0,
         4096, 1024, 4096, 1.f, 0, 1, 1, 0, 0, 0, 0, 0, 0);
    ln_kernel<<<dim3(4096), dim3(256), 0, stream>>>(X32, Tmp, ln2s + L * 1024,
                                                    ln2b + L * 1024, X32, Xh);
  }

  // final LN + head on the 8 last-token rows
  ln_kernel<<<dim3(4096), dim3(256), 0, stream>>>(X32, nullptr, lnfs, lnfb, X32, Xh);
  head_kernel<<<dim3(8, 2), dim3(256), 0, stream>>>(X32, Wh, bh, out);
}
